// RPEConformerBlock_89189290869053
// MI455X (gfx1250) — matrix operand order repaired and checked by reading, not compiled, not run on hardware
//
#include <hip/hip_runtime.h>
#include <hip/hip_bf16.h>
#include <cstdint>
#include <cstddef>

// ---------------------------------------------------------------------------
// Conformer block forward for MI455X (gfx1250, wave32, WMMA).
// All GEMM-shaped work lowers to v_wmma_f32_16x16x32_bf16.
// Weights are pre-transposed (N x K, bf16) once per launch so every WMMA
// B-fragment is a contiguous 32B global load; the GEMM inner loop is
// software-pipelined (load k+1 fragments while k's WMMAs issue) so the four
// WMMAs per step are independent and back-to-back.
// ---------------------------------------------------------------------------

typedef __attribute__((ext_vector_type(16))) __bf16 bf16x16;
typedef __attribute__((ext_vector_type(8)))  __bf16 bf16x8;
typedef __attribute__((ext_vector_type(8)))  float  f32x8;

#define NB   4
#define TT   1024
#define DD   512
#define HH   8
#define DH   64
#define FFN  2048
#define KW   31
#define MM   (NB * TT)      // 4096 rows

// ---- scalar bf16 helpers (explicit RNE, no implicit __bf16 arithmetic) ----
__device__ __forceinline__ __bf16 f2bf(float f) {
    unsigned int u = __builtin_bit_cast(unsigned int, f);
    u += 0x7FFFu + ((u >> 16) & 1u);
    unsigned short s = (unsigned short)(u >> 16);
    return __builtin_bit_cast(__bf16, s);
}
__device__ __forceinline__ float bf2f(__bf16 x) {
    unsigned short s = __builtin_bit_cast(unsigned short, x);
    unsigned int u = ((unsigned int)s) << 16;
    return __builtin_bit_cast(float, u);
}
__device__ __forceinline__ float silu(float v) {
    return v * (1.0f / (1.0f + __expf(-v)));
}

// ---- WMMA fragment builders in contiguous k order (k = 16*half + s, A and B alike) -------------------
// A (16x32): lane = row M (lanes 16-31 mirror M with the other K half):
//   lanes 0-15 : k = [0..15];  lanes 16-31: k = [16..31]
__device__ __forceinline__ bf16x16 make_afrag(const __bf16* row, int half) {
    bf16x8 lo = *(const bf16x8*)(row + half * 16);
    bf16x8 hi = *(const bf16x8*)(row + half * 16 + 8);
    bf16x16 r;
#pragma unroll
    for (int i = 0; i < 8; ++i) { r[i] = lo[i]; r[8 + i] = hi[i]; }
    return r;
}
// B (32x16): lane<16 holds column n=lane, k=[0..15] packed 2/VGPR; lanes 16-31
// hold the same column n=lane-16 with k=[16..31]  -> one contiguous 32B read
// per lane when B^T (n-major, k contiguous) is in memory.
__device__ __forceinline__ f32x8 wmma_bf16(bf16x16 a, bf16x16 b, f32x8 c) {
    return __builtin_amdgcn_wmma_f32_16x16x32_bf16(false, a, false, b,
                                                   (short)0, c, false, false);
}

// ---------------------------------------------------------------------------
// GEMM (B pre-transposed): C[M,N] = epilogue(A[M,K] @ Bt[N,K]^T + bias)
// block = 256 threads = 8 waves; each wave owns a 16(M) x 64(N) tile
// (4 accumulators, A fragment reused 4x). Block macro-tile 16 x 512.
// Software-pipelined: fragments for k+32 are loaded before the k WMMAs.
// Requires M%16==0, N%512==0, K%32==0 (true for all uses here).
// act: 0 = none, 1 = SiLU.  out = resid + scale*act(acc + bias)
// ---------------------------------------------------------------------------
__global__ void __launch_bounds__(256)
gemm_bf16_nt(const __bf16* __restrict__ A, const __bf16* __restrict__ Bt,
             const float* __restrict__ bias, const float* __restrict__ resid,
             float* __restrict__ Cf, __bf16* __restrict__ Cb,
             int M, int N, int K, float scale, int act)
{
    const int tid  = threadIdx.x;
    const int wave = tid >> 5;
    const int lane = tid & 31;
    const int l15  = lane & 15;
    const int half = lane >> 4;
    const int m0   = blockIdx.y * 16;
    const int nb   = blockIdx.x * 512 + wave * 64;

    f32x8 acc0 = {0.f, 0.f, 0.f, 0.f, 0.f, 0.f, 0.f, 0.f};
    f32x8 acc1 = acc0, acc2 = acc0, acc3 = acc0;

    const __bf16* arow = A + (size_t)(m0 + l15) * K;
    const __bf16* b0r  = Bt + (size_t)(nb + 0 * 16 + l15) * K + half * 16;
    const __bf16* b1r  = Bt + (size_t)(nb + 1 * 16 + l15) * K + half * 16;
    const __bf16* b2r  = Bt + (size_t)(nb + 2 * 16 + l15) * K + half * 16;
    const __bf16* b3r  = Bt + (size_t)(nb + 3 * 16 + l15) * K + half * 16;

    // prologue: fragments for k = 0
    bf16x16 af = make_afrag(arow, half);
    bf16x16 f0 = *(const bf16x16*)(b0r);
    bf16x16 f1 = *(const bf16x16*)(b1r);
    bf16x16 f2 = *(const bf16x16*)(b2r);
    bf16x16 f3 = *(const bf16x16*)(b3r);

    for (int k0 = 0; k0 < K - 32; k0 += 32) {
        const int kn = k0 + 32;
        // hint k+2 into cache (global_prefetch_b8 path)
        __builtin_prefetch((const void*)(arow + kn + 32), 0, 1);
        __builtin_prefetch((const void*)(b0r + kn + 32), 0, 1);
        // load k+1 fragments into fresh registers
        bf16x16 an = make_afrag(arow + kn, half);
        bf16x16 g0 = *(const bf16x16*)(b0r + kn);
        bf16x16 g1 = *(const bf16x16*)(b1r + kn);
        bf16x16 g2 = *(const bf16x16*)(b2r + kn);
        bf16x16 g3 = *(const bf16x16*)(b3r + kn);
        // four independent back-to-back WMMAs on the k fragments
        acc0 = wmma_bf16(af, f0, acc0);
        acc1 = wmma_bf16(af, f1, acc1);
        acc2 = wmma_bf16(af, f2, acc2);
        acc3 = wmma_bf16(af, f3, acc3);
        af = an; f0 = g0; f1 = g1; f2 = g2; f3 = g3;
    }
    acc0 = wmma_bf16(af, f0, acc0);
    acc1 = wmma_bf16(af, f1, acc1);
    acc2 = wmma_bf16(af, f2, acc2);
    acc3 = wmma_bf16(af, f3, acc3);

    // Epilogue. C layout: VGPR r -> M = r + 8*half, N = lane&15.
    f32x8 accs[4] = {acc0, acc1, acc2, acc3};
#pragma unroll
    for (int j = 0; j < 4; ++j) {
        const int n = nb + j * 16 + l15;
        float bval = bias ? bias[n] : 0.0f;
#pragma unroll
        for (int r = 0; r < 8; ++r) {
            int m = m0 + r + half * 8;
            float vv = accs[j][r] + bval;
            if (act == 1) vv = silu(vv);
            vv *= scale;
            if (resid) vv += resid[(size_t)m * N + n];
            if (Cf) Cf[(size_t)m * N + n] = vv;
            if (Cb) Cb[(size_t)m * N + n] = f2bf(vv);
        }
    }
}

// ---------------------------------------------------------------------------
// Attention scores with fused Transformer-XL rel-shift:
//   S[n,h,t,s] = ( Qu[t].K[s] + Qv[t].R[t-s+T-1] ) * dh^-0.5   (bf16 out)
// One wave per 16x16 (t,s) tile. AC via 2 WMMAs; BD via 2 WMMA tiles of
// BD_raw over p = pa..pa+31 (pa = t0-s0+T-16 >= 0, pa+31 <= 2047 with the
// padded/zeroed R row 2047), staged to LDS and diagonally gathered.
// ---------------------------------------------------------------------------
__global__ void __launch_bounds__(256)
attn_scores_kernel(const __bf16* __restrict__ Qu, const __bf16* __restrict__ Qv,
                   const __bf16* __restrict__ Kmat, const __bf16* __restrict__ R,
                   __bf16* __restrict__ S)
{
    __shared__ float bd[8][16][32];
    const int tid  = threadIdx.x;
    const int wave = tid >> 5;
    const int lane = tid & 31;
    const int l15  = lane & 15;
    const int half = lane >> 4;

    const int nh = blockIdx.z;            // n*H + h
    const int n  = nh >> 3;
    const int h  = nh & 7;
    const int t0 = blockIdx.y * 16;
    const int s0 = (blockIdx.x * 8 + wave) * 16;
    const int pa = t0 - s0 + TT - 16;     // >= 0 always

    const size_t qrow = ((size_t)n * TT + t0 + l15) * DD + h * DH;
    const size_t krow = ((size_t)n * TT + s0 + l15) * DD + h * DH;

    f32x8 ac = {0.f, 0.f, 0.f, 0.f, 0.f, 0.f, 0.f, 0.f};
    f32x8 b0 = ac, b1 = ac;

#pragma unroll
    for (int k0 = 0; k0 < DH; k0 += 32) {
        bf16x16 aqu = make_afrag(Qu + qrow + k0, half);
        bf16x16 aqv = make_afrag(Qv + qrow + k0, half);
        // K is (s, d) row-major == (K^T)^T : exactly the "nt" fragment pattern
        bf16x16 bk = *(const bf16x16*)(Kmat + krow + k0 + half * 16);
        ac = wmma_bf16(aqu, bk, ac);
        bf16x16 br0 = *(const bf16x16*)(R + ((size_t)(pa + l15)) * DD + h * DH + k0 + half * 16);
        bf16x16 br1 = *(const bf16x16*)(R + ((size_t)(pa + 16 + l15)) * DD + h * DH + k0 + half * 16);
        b0 = wmma_bf16(aqv, br0, b0);
        b1 = wmma_bf16(aqv, br1, b1);
    }

    // stage BD_raw tile (16 x 32) to LDS (wave-private slice)
#pragma unroll
    for (int r = 0; r < 8; ++r) {
        int mi = r + half * 8;
        bd[wave][mi][l15]      = b0[r];
        bd[wave][mi][16 + l15] = b1[r];
    }
    asm volatile("s_wait_dscnt 0" ::: "memory");   // wave-local LDS RAW

#pragma unroll
    for (int r = 0; r < 8; ++r) {
        int mi = r + half * 8;
        int j  = l15;
        float vv = (ac[r] + bd[wave][mi][mi - j + 15]) * 0.125f; // dh^-0.5
        S[((size_t)nh * TT + t0 + mi) * TT + s0 + j] = f2bf(vv);
    }
}

// ---------------------------------------------------------------------------
// Row softmax in place (bf16 rows of length T=1024), f32 math.
// ---------------------------------------------------------------------------
__global__ void __launch_bounds__(256)
softmax_kernel(__bf16* __restrict__ S)
{
    __shared__ float red[256];
    const int tid = threadIdx.x;
    __bf16* p = S + (size_t)blockIdx.x * TT;

    float v[4];
    float mx = -3.0e38f;
#pragma unroll
    for (int i = 0; i < 4; ++i) {
        v[i] = bf2f(p[tid + i * 256]);
        mx = fmaxf(mx, v[i]);
    }
    red[tid] = mx; __syncthreads();
    for (int s = 128; s > 0; s >>= 1) {
        if (tid < s) red[tid] = fmaxf(red[tid], red[tid + s]);
        __syncthreads();
    }
    mx = red[0]; __syncthreads();

    float sum = 0.f;
#pragma unroll
    for (int i = 0; i < 4; ++i) { v[i] = __expf(v[i] - mx); sum += v[i]; }
    red[tid] = sum; __syncthreads();
    for (int s = 128; s > 0; s >>= 1) {
        if (tid < s) red[tid] += red[tid + s];
        __syncthreads();
    }
    float inv = 1.0f / red[0];
#pragma unroll
    for (int i = 0; i < 4; ++i) p[tid + i * 256] = f2bf(v[i] * inv);
}

// ---------------------------------------------------------------------------
// AV: out[n,t,h*64+d] = sum_s P[n,h,t,s] * V[n,h,s,d]
// V pre-transposed to Vt[(n*H+h)*64+d][t] so B fragments are contiguous.
// Block = 256 threads = 8 waves; each wave owns one full 16(t) x 64(d) tile
// (4 accumulators, probs A-fragment loaded once per k-step and reused 4x).
// Software-pipelined like the main GEMM.
// ---------------------------------------------------------------------------
__global__ void __launch_bounds__(256)
attn_av_kernel(const __bf16* __restrict__ P, const __bf16* __restrict__ Vt,
               __bf16* __restrict__ O)
{
    const int tid  = threadIdx.x;
    const int wave = tid >> 5;
    const int lane = tid & 31;
    const int l15  = lane & 15;
    const int half = lane >> 4;

    const int t0 = (blockIdx.x * 8 + wave) * 16;
    const int nh = blockIdx.y;
    const int n  = nh >> 3;
    const int h  = nh & 7;

    const __bf16* prow = P + ((size_t)nh * TT + t0 + l15) * TT;
    const __bf16* v0r  = Vt + ((size_t)nh * DH + 0 * 16 + l15) * TT + half * 16;
    const __bf16* v1r  = Vt + ((size_t)nh * DH + 1 * 16 + l15) * TT + half * 16;
    const __bf16* v2r  = Vt + ((size_t)nh * DH + 2 * 16 + l15) * TT + half * 16;
    const __bf16* v3r  = Vt + ((size_t)nh * DH + 3 * 16 + l15) * TT + half * 16;

    f32x8 acc0 = {0.f, 0.f, 0.f, 0.f, 0.f, 0.f, 0.f, 0.f};
    f32x8 acc1 = acc0, acc2 = acc0, acc3 = acc0;

    bf16x16 af = make_afrag(prow, half);
    bf16x16 f0 = *(const bf16x16*)(v0r);
    bf16x16 f1 = *(const bf16x16*)(v1r);
    bf16x16 f2 = *(const bf16x16*)(v2r);
    bf16x16 f3 = *(const bf16x16*)(v3r);

    for (int k0 = 0; k0 < TT - 32; k0 += 32) {
        const int kn = k0 + 32;
        __builtin_prefetch((const void*)(prow + kn + 32), 0, 1);
        bf16x16 an = make_afrag(prow + kn, half);
        bf16x16 g0 = *(const bf16x16*)(v0r + kn);
        bf16x16 g1 = *(const bf16x16*)(v1r + kn);
        bf16x16 g2 = *(const bf16x16*)(v2r + kn);
        bf16x16 g3 = *(const bf16x16*)(v3r + kn);
        acc0 = wmma_bf16(af, f0, acc0);
        acc1 = wmma_bf16(af, f1, acc1);
        acc2 = wmma_bf16(af, f2, acc2);
        acc3 = wmma_bf16(af, f3, acc3);
        af = an; f0 = g0; f1 = g1; f2 = g2; f3 = g3;
    }
    acc0 = wmma_bf16(af, f0, acc0);
    acc1 = wmma_bf16(af, f1, acc1);
    acc2 = wmma_bf16(af, f2, acc2);
    acc3 = wmma_bf16(af, f3, acc3);

    f32x8 accs[4] = {acc0, acc1, acc2, acc3};
#pragma unroll
    for (int j = 0; j < 4; ++j) {
        const int d = j * 16 + l15;
#pragma unroll
        for (int r = 0; r < 8; ++r) {
            int t = t0 + r + half * 8;
            O[(((size_t)n * TT + t) * DD) + h * DH + d] = f2bf(accs[j][r]);
        }
    }
}

// ---------------------------------------------------------------------------
// LayerNorm over D=512; one block (256 thr) per row; biased variance (jnp.var).
// ---------------------------------------------------------------------------
__global__ void __launch_bounds__(256)
layernorm512(const float* __restrict__ X, const float* __restrict__ g,
             const float* __restrict__ b, __bf16* __restrict__ o16,
             float* __restrict__ o32)
{
    __shared__ float red[256];
    const int tid = threadIdx.x;
    const float* x = X + (size_t)blockIdx.x * DD;
    float a0 = x[tid], a1 = x[tid + 256];

    red[tid] = a0 + a1; __syncthreads();
    for (int s = 128; s > 0; s >>= 1) {
        if (tid < s) red[tid] += red[tid + s];
        __syncthreads();
    }
    float mean = red[0] * (1.0f / DD); __syncthreads();

    float d0 = a0 - mean, d1 = a1 - mean;
    red[tid] = d0 * d0 + d1 * d1; __syncthreads();
    for (int s = 128; s > 0; s >>= 1) {
        if (tid < s) red[tid] += red[tid + s];
        __syncthreads();
    }
    float rstd = rsqrtf(red[0] * (1.0f / DD) + 1e-5f);

    float y0 = d0 * rstd * g[tid] + b[tid];
    float y1 = d1 * rstd * g[tid + 256] + b[tid + 256];
    size_t base = (size_t)blockIdx.x * DD;
    if (o16) { o16[base + tid] = f2bf(y0); o16[base + tid + 256] = f2bf(y1); }
    if (o32) { o32[base + tid] = y0;       o32[base + tid + 256] = y1; }
}

// ---------------------------------------------------------------------------
// Small elementwise kernels
// ---------------------------------------------------------------------------
__global__ void f2b_kernel(const float* __restrict__ src, __bf16* __restrict__ dst,
                           int n_src, int n_dst)
{
    int i = blockIdx.x * blockDim.x + threadIdx.x;
    if (i < n_dst) dst[i] = (i < n_src) ? f2bf(src[i]) : f2bf(0.0f);
}

// convert f32 (K x N) weight -> bf16 transposed (N x K)
__global__ void f2b_t_kernel(const float* __restrict__ src, __bf16* __restrict__ dst,
                             int K, int N, int total)
{
    int i = blockIdx.x * blockDim.x + threadIdx.x;
    if (i >= total) return;
    int k = i / N, n = i - k * N;          // coalesced read along n
    dst[(size_t)n * K + k] = f2bf(src[i]);
}

__global__ void copy_f32(const float* __restrict__ src, float* __restrict__ dst, int n)
{
    int i = blockIdx.x * blockDim.x + threadIdx.x;
    if (i < n) dst[i] = src[i];
}

__global__ void add_uv_kernel(const float* __restrict__ Q, const float* __restrict__ u,
                              const float* __restrict__ v, __bf16* __restrict__ Qu,
                              __bf16* __restrict__ Qv, int total)
{
    int i = blockIdx.x * blockDim.x + threadIdx.x;
    if (i >= total) return;
    int c = i & (DD - 1);
    float q = Q[i];
    Qu[i] = f2bf(q + u[c]);
    Qv[i] = f2bf(q + v[c]);
}

__global__ void make_vt_kernel(const float* __restrict__ Vf, __bf16* __restrict__ Vt,
                               int total)
{
    int i = blockIdx.x * blockDim.x + threadIdx.x;
    if (i >= total) return;
    int c = i & (DD - 1);
    int m = i >> 9;
    int n = m >> 10, t = m & (TT - 1);
    int h = c >> 6, d = c & (DH - 1);
    Vt[(((size_t)(n * HH + h)) * DH + d) * TT + t] = f2bf(Vf[i]);
}

__global__ void glu_kernel(const float* __restrict__ C1, float* __restrict__ G, int total)
{
    int i = blockIdx.x * blockDim.x + threadIdx.x;
    if (i >= total) return;
    int c = i & (DD - 1);
    int m = i >> 9;
    float a = C1[(size_t)m * (2 * DD) + c];
    float g = C1[(size_t)m * (2 * DD) + DD + c];
    G[i] = a * (1.0f / (1.0f + __expf(-g)));
}

// depthwise conv (K=31, pad 15) + dw_b + BN(eval) + SiLU, bf16 out
__global__ void dwconv_kernel(const float* __restrict__ G, const float* __restrict__ W,
                              const float* __restrict__ db, const float* __restrict__ bg,
                              const float* __restrict__ bb, __bf16* __restrict__ O,
                              int total)
{
    int i = blockIdx.x * blockDim.x + threadIdx.x;
    if (i >= total) return;
    int c = i & (DD - 1);
    int m = i >> 9;
    int n = m >> 10, t = m & (TT - 1);
    float s = 0.f;
#pragma unroll
    for (int k = 0; k < KW; ++k) {
        int tt = t + k - (KW / 2);
        if (tt >= 0 && tt < TT)
            s += G[((size_t)(n * TT + tt)) * DD + c] * W[c * KW + k];
    }
    s += db[c];
    s = s * bg[c] + bb[c];
    O[i] = f2bf(silu(s));
}

// ---------------------------------------------------------------------------
// Host orchestration
// ---------------------------------------------------------------------------
extern "C" void kernel_launch(void* const* d_in, const int* in_sizes, int n_in,
                              void* d_out, int out_size, void* d_ws, size_t ws_size,
                              hipStream_t stream)
{
    (void)in_sizes; (void)n_in; (void)out_size; (void)ws_size;
    const float* x      = (const float*)d_in[0];
    const float* rel_pe = (const float*)d_in[1];
    const float* ln1_g  = (const float*)d_in[2];
    const float* ln1_b  = (const float*)d_in[3];
    const float* ff1_w1 = (const float*)d_in[4];
    const float* ff1_b1 = (const float*)d_in[5];
    const float* ff1_w2 = (const float*)d_in[6];
    const float* ff1_b2 = (const float*)d_in[7];
    const float* an_g   = (const float*)d_in[8];
    const float* an_b   = (const float*)d_in[9];
    const float* wq     = (const float*)d_in[10];
    const float* bq     = (const float*)d_in[11];
    const float* wk     = (const float*)d_in[12];
    const float* bk     = (const float*)d_in[13];
    const float* wv     = (const float*)d_in[14];
    const float* bv     = (const float*)d_in[15];
    const float* wo     = (const float*)d_in[16];
    const float* bo     = (const float*)d_in[17];
    const float* pos_w  = (const float*)d_in[18];
    const float* uvecp  = (const float*)d_in[19];
    const float* vvecp  = (const float*)d_in[20];
    const float* cn_g   = (const float*)d_in[21];
    const float* cn_b   = (const float*)d_in[22];
    const float* pe_w   = (const float*)d_in[23];
    const float* pe_b   = (const float*)d_in[24];
    const float* dw_w   = (const float*)d_in[25];
    const float* dw_b   = (const float*)d_in[26];
    const float* bn_g   = (const float*)d_in[27];
    const float* bn_b   = (const float*)d_in[28];
    const float* pc_w   = (const float*)d_in[29];
    const float* pc_b   = (const float*)d_in[30];
    const float* ln2_g  = (const float*)d_in[31];
    const float* ln2_b  = (const float*)d_in[32];
    const float* ff2_w1 = (const float*)d_in[33];
    const float* ff2_b1 = (const float*)d_in[34];
    const float* ff2_w2 = (const float*)d_in[35];
    const float* ff2_b2 = (const float*)d_in[36];
    const float* fn_g   = (const float*)d_in[37];
    const float* fn_b   = (const float*)d_in[38];

    // ---- workspace cursor allocation (256B aligned) ----
    char* base = (char*)d_ws;
    size_t off = 0;
    auto alloc = [&](size_t bytes) -> void* {
        void* p = base + off;
        off += (bytes + 255) & ~(size_t)255;
        return p;
    };
    auto abf = [&](size_t elems) -> __bf16* { return (__bf16*)alloc(elems * 2); };
    auto af32 = [&](size_t elems) -> float* { return (float*)alloc(elems * 4); };

    // all weight buffers hold the TRANSPOSED (N x K) bf16 weight
    __bf16* w1a16  = abf((size_t)DD * FFN);       // (FFN x DD)
    __bf16* w1b16  = abf((size_t)FFN * DD);       // (DD x FFN)
    __bf16* wq16   = abf((size_t)DD * DD);
    __bf16* wk16   = abf((size_t)DD * DD);
    __bf16* wv16   = abf((size_t)DD * DD);
    __bf16* wo16   = abf((size_t)DD * DD);
    __bf16* posw16 = abf((size_t)DD * DD);
    __bf16* pew16  = abf((size_t)DD * 2 * DD);    // (2DD x DD)
    __bf16* pcw16  = abf((size_t)DD * DD);
    __bf16* w2a16  = abf((size_t)DD * FFN);
    __bf16* w2b16  = abf((size_t)FFN * DD);
    __bf16* rel16  = abf((size_t)2048 * DD);      // rel_pe padded 2047 -> 2048
    float*  xf     = af32((size_t)MM * DD);       // running residual x
    __bf16* ln16   = abf((size_t)MM * DD);        // reused LN output
    __bf16* big16  = abf((size_t)MM * FFN);       // H1 / c1(f32 view) / H2
    float*  bigf   = (float*)big16;               // M x 1024 f32 (same bytes)
    float*  qf     = af32((size_t)MM * DD);
    float*  vf     = af32((size_t)MM * DD);
    __bf16* qu16   = abf((size_t)MM * DD);
    __bf16* qv16   = abf((size_t)MM * DD);
    __bf16* k16    = abf((size_t)MM * DD);
    __bf16* vt16   = abf((size_t)NB * HH * DH * TT);
    __bf16* r16    = abf((size_t)2048 * DD);
    __bf16* s16    = abf((size_t)NB * HH * TT * TT); // scores / probs (64MB)
    __bf16* attn16 = abf((size_t)MM * DD);
    float*  gluf   = af32((size_t)MM * DD);
    __bf16* conv16 = abf((size_t)MM * DD);

    auto cvt = [&](const float* s, __bf16* d, int n_src, int n_dst) {
        f2b_kernel<<<(n_dst + 255) / 256, 256, 0, stream>>>(s, d, n_src, n_dst);
    };
    auto cvtT = [&](const float* s, __bf16* d, int K, int N) {
        int total = K * N;
        f2b_t_kernel<<<(total + 255) / 256, 256, 0, stream>>>(s, d, K, N, total);
    };
    auto gemm = [&](const __bf16* A, const __bf16* Bt, const float* bias,
                    const float* resid, float* Cf, __bf16* Cb,
                    int Mm, int Nn, int Kk, float scale, int act) {
        dim3 g(Nn / 512, Mm / 16);
        gemm_bf16_nt<<<g, 256, 0, stream>>>(A, Bt, bias, resid, Cf, Cb,
                                            Mm, Nn, Kk, scale, act);
    };
    auto ln = [&](const float* X, const float* g, const float* b,
                  __bf16* o16, float* o32) {
        layernorm512<<<MM, 256, 0, stream>>>(X, g, b, o16, o32);
    };

    // ---- weight conversions (bf16, transposed to N x K) ----
    cvtT(ff1_w1, w1a16, DD, FFN);
    cvtT(ff1_w2, w1b16, FFN, DD);
    cvtT(wq, wq16, DD, DD);
    cvtT(wk, wk16, DD, DD);
    cvtT(wv, wv16, DD, DD);
    cvtT(wo, wo16, DD, DD);
    cvtT(pos_w, posw16, DD, DD);
    cvtT(pe_w, pew16, DD, 2 * DD);
    cvtT(pc_w, pcw16, DD, DD);
    cvtT(ff2_w1, w2a16, DD, FFN);
    cvtT(ff2_w2, w2b16, FFN, DD);
    cvt(rel_pe, rel16, (2 * TT - 1) * DD, 2048 * DD);   // zero-pad row 2047

    copy_f32<<<(MM * DD + 255) / 256, 256, 0, stream>>>(x, xf, MM * DD);

    // ---- macaron FF 1 (half step) ----
    ln(xf, ln1_g, ln1_b, ln16, nullptr);
    gemm(ln16, w1a16, ff1_b1, nullptr, nullptr, big16, MM, FFN, DD, 1.0f, 1);
    gemm(big16, w1b16, ff1_b2, xf, xf, nullptr, MM, DD, FFN, 0.5f, 0);

    // ---- RPE multi-head attention ----
    ln(xf, an_g, an_b, ln16, nullptr);
    gemm(ln16, wq16, bq, nullptr, qf, nullptr, MM, DD, DD, 1.0f, 0);
    gemm(ln16, wk16, bk, nullptr, nullptr, k16, MM, DD, DD, 1.0f, 0);
    gemm(ln16, wv16, bv, nullptr, vf, nullptr, MM, DD, DD, 1.0f, 0);
    add_uv_kernel<<<(MM * DD + 255) / 256, 256, 0, stream>>>(qf, uvecp, vvecp,
                                                             qu16, qv16, MM * DD);
    make_vt_kernel<<<(MM * DD + 255) / 256, 256, 0, stream>>>(vf, vt16, MM * DD);
    gemm(rel16, posw16, nullptr, nullptr, nullptr, r16, 2048, DD, DD, 1.0f, 0);

    attn_scores_kernel<<<dim3(TT / 16 / 8, TT / 16, NB * HH), 256, 0, stream>>>(
        qu16, qv16, k16, r16, s16);
    softmax_kernel<<<NB * HH * TT, 256, 0, stream>>>(s16);
    attn_av_kernel<<<dim3(TT / 16 / 8, NB * HH), 256, 0, stream>>>(s16, vt16, attn16);
    gemm(attn16, wo16, bo, xf, xf, nullptr, MM, DD, DD, 1.0f, 0);

    // ---- conv module ----
    ln(xf, cn_g, cn_b, ln16, nullptr);
    gemm(ln16, pew16, pe_b, nullptr, bigf, nullptr, MM, 2 * DD, DD, 1.0f, 0);
    glu_kernel<<<(MM * DD + 255) / 256, 256, 0, stream>>>(bigf, gluf, MM * DD);
    dwconv_kernel<<<(MM * DD + 255) / 256, 256, 0, stream>>>(gluf, dw_w, dw_b,
                                                             bn_g, bn_b, conv16,
                                                             MM * DD);
    gemm(conv16, pcw16, pc_b, xf, xf, nullptr, MM, DD, DD, 1.0f, 0);

    // ---- macaron FF 2 + final LN ----
    ln(xf, ln2_g, ln2_b, ln16, nullptr);
    gemm(ln16, w2a16, ff2_b1, nullptr, nullptr, big16, MM, FFN, DD, 1.0f, 1);
    gemm(big16, w2b16, ff2_b2, xf, xf, nullptr, MM, DD, FFN, 0.5f, 0);
    ln(xf, fn_g, fn_b, nullptr, (float*)d_out);
}
